// CrossModalFusion_17394617549498
// MI455X (gfx1250) — matrix operand order repaired and checked by reading, not compiled, not run on hardware
//
#include <hip/hip_runtime.h>
#include <hip/hip_bf16.h>
#include <stddef.h>
#include <stdint.h>

// ---------------------------------------------------------------------------
// CDNA5 (gfx1250, wave32) fused cross-modal attention + LayerNorm.
// All GEMMs run on v_wmma_f32_16x16x32_bf16 (fp32 accumulation).
// bf16->LDS staging uses GLOBAL_LOAD_ASYNC_TO_LDS_B128 when available.
// ---------------------------------------------------------------------------

typedef __attribute__((ext_vector_type(16))) __bf16 v16bf;
typedef __attribute__((ext_vector_type(8)))  __bf16 v8bf;
typedef __attribute__((ext_vector_type(8)))  float  v8f;
typedef int v4i __attribute__((vector_size(16)));

#define HID 512
#define DIM 400
#define SEQ 2048
#define BATCH 32

#if defined(__has_builtin)
#if __has_builtin(__builtin_amdgcn_global_load_async_to_lds_b128) && \
    __has_builtin(__builtin_amdgcn_s_wait_asynccnt)
#define ATHENA_ASYNC 1
#endif
#endif
#ifndef ATHENA_ASYNC
#define ATHENA_ASYNC 0
#endif

// 16-byte global->LDS copy: async DMA path on gfx1250, VGPR fallback otherwise.
__device__ __forceinline__ void cp16(__bf16* dst_lds, const __bf16* src_glob) {
#if ATHENA_ASYNC
    __builtin_amdgcn_global_load_async_to_lds_b128(
        (__attribute__((address_space(1))) v4i*)src_glob,
        (__attribute__((address_space(3))) v4i*)dst_lds, 0, 0);
#else
    *(uint4*)dst_lds = *(const uint4*)src_glob;
#endif
}

__device__ __forceinline__ void async_join() {
#if ATHENA_ASYNC
    __builtin_amdgcn_s_wait_asynccnt(0);
#endif
}

__device__ __forceinline__ v8f zero8() {
    v8f z;
#pragma unroll
    for (int i = 0; i < 8; ++i) z[i] = 0.0f;
    return z;
}

// A-matrix fragment (16x32, 16-bit), contiguous K order matching load_b_frag:
// lanes 0-15: M=lane, K 0..15; lanes 16-31: M=lane-16, K 16..31
__device__ __forceinline__ v16bf load_a_frag(const __bf16* A, int ld, int lane) {
    const int row = lane & 15;
    const int kb  = (lane >> 4) << 4;   // 0 or 16
    const __bf16* p = A + (size_t)row * ld;
    v16bf a;
#pragma unroll
    for (int i = 0; i < 8; ++i) {
        a[i]     = p[kb + i];
        a[i + 8] = p[kb + 8 + i];
    }
    return a;
}

// B-matrix fragment (32x16, 16-bit): lane = N column; lanes 0-15 hold K=0..15,
// lanes 16-31 hold K=16..31.  Tiles are staged n-major so the 16 K-values are
// contiguous in LDS.
__device__ __forceinline__ v16bf load_b_frag(const __bf16* p) {
    v8bf lo = *(const v8bf*)p;
    v8bf hi = *(const v8bf*)(p + 8);
    v16bf b;
#pragma unroll
    for (int i = 0; i < 8; ++i) { b[i] = lo[i]; b[i + 8] = hi[i]; }
    return b;
}

__device__ __forceinline__ v8f wmma_bf16(v16bf a, v16bf b, v8f c) {
    return __builtin_amdgcn_wmma_f32_16x16x32_bf16(false, a, false, b, (short)0, c,
                                                   false, false);
}

// convert 8 consecutive floats -> 8 bf16 in LDS
__device__ __forceinline__ void cvt_store8(__bf16* d, const float* s) {
    const float4 f0 = *(const float4*)s;
    const float4 f1 = *(const float4*)(s + 4);
    d[0] = (__bf16)f0.x; d[1] = (__bf16)f0.y; d[2] = (__bf16)f0.z; d[3] = (__bf16)f0.w;
    d[4] = (__bf16)f1.x; d[5] = (__bf16)f1.y; d[6] = (__bf16)f1.z; d[7] = (__bf16)f1.w;
}

// ---------------------------------------------------------------------------
// Kernel 1: out[M,512] = bf16(in[M,400] @ W[400,512] + bias)   (Q, K, V)
// grid: (M/64, 512/128), block 128 (4 waves, 16 rows each)
// ---------------------------------------------------------------------------
__global__ __launch_bounds__(128) void qkv_kernel(
    const float* __restrict__ in, const float* __restrict__ W,
    const float* __restrict__ bias, __bf16* __restrict__ out) {
    extern __shared__ char smem[];
    __bf16* As = (__bf16*)smem;                 // [64][32] row-major
    __bf16* Bs = (__bf16*)(smem + 64 * 32 * 2); // [128][32] n-major (transposed)

    const int tid  = threadIdx.x;
    const int lane = tid & 31;
    const int wv   = tid >> 5;
    const int row0 = blockIdx.x * 64;
    const int col0 = blockIdx.y * 128;

    v8f acc[8];
#pragma unroll
    for (int t = 0; t < 8; ++t) acc[t] = zero8();

    for (int ks = 0; ks < 13; ++ks) {           // ceil(400/32) with zero-pad
        const int k0 = ks * 32;
        // A tile: 64x32, staged in 8-element chunks (2 chunks/thread)
        for (int c = tid; c < 256; c += 128) {
            const int r = c >> 2, ch = c & 3;
            const int k = k0 + ch * 8;
            __bf16* d = As + r * 32 + ch * 8;
            if (k + 8 <= DIM) {
                cvt_store8(d, in + (size_t)(row0 + r) * DIM + k);
            } else {
#pragma unroll
                for (int j = 0; j < 8; ++j)
                    d[j] = (__bf16)((k + j < DIM) ? in[(size_t)(row0 + r) * DIM + k + j]
                                                  : 0.0f);
            }
        }
        // B tile: 128 cols x 32 k, coalesced reads of 8 columns, n-major stores
        for (int c = tid; c < 512; c += 128) {
            const int kk = c & 31;
            const int n0 = (c >> 5) * 8;
            const int k  = k0 + kk;
            if (k < DIM) {
                const float4 f0 = *(const float4*)(W + (size_t)k * HID + col0 + n0);
                const float4 f1 = *(const float4*)(W + (size_t)k * HID + col0 + n0 + 4);
                Bs[(n0 + 0) * 32 + kk] = (__bf16)f0.x;
                Bs[(n0 + 1) * 32 + kk] = (__bf16)f0.y;
                Bs[(n0 + 2) * 32 + kk] = (__bf16)f0.z;
                Bs[(n0 + 3) * 32 + kk] = (__bf16)f0.w;
                Bs[(n0 + 4) * 32 + kk] = (__bf16)f1.x;
                Bs[(n0 + 5) * 32 + kk] = (__bf16)f1.y;
                Bs[(n0 + 6) * 32 + kk] = (__bf16)f1.z;
                Bs[(n0 + 7) * 32 + kk] = (__bf16)f1.w;
            } else {
#pragma unroll
                for (int j = 0; j < 8; ++j) Bs[(n0 + j) * 32 + kk] = (__bf16)0.0f;
            }
        }
        __syncthreads();

        const v16bf a = load_a_frag(As + wv * 16 * 32, 32, lane);
#pragma unroll
        for (int t = 0; t < 8; ++t) {
            const v16bf b = load_b_frag(Bs + (t * 16 + (lane & 15)) * 32 + ((lane >> 4) * 16));
            acc[t] = wmma_bf16(a, b, acc[t]);
        }
        __syncthreads();
    }

    const int rbase = row0 + wv * 16 + ((lane >> 4) ? 8 : 0);
#pragma unroll
    for (int t = 0; t < 8; ++t) {
        const int col = col0 + t * 16 + (lane & 15);
        const float bv = bias[col];
#pragma unroll
        for (int r = 0; r < 8; ++r)
            out[(size_t)(rbase + r) * HID + col] = (__bf16)(acc[t][r] + bv);
    }
}

// ---------------------------------------------------------------------------
// Kernel 2: flash attention.  O = softmax(Q K^T / sqrt(512)) V, bf16 in/out.
// grid: (SEQ/64, BATCH), block 128 (4 waves; each wave owns 16 q-rows, H=512)
// LDS: Qs[64][512] + Ks[32][512] + Vt[512][32] + P scratch  = 132 KB
// ---------------------------------------------------------------------------
__global__ __launch_bounds__(128, 1) void attn_kernel(
    const __bf16* __restrict__ Qb, const __bf16* __restrict__ Kb,
    const __bf16* __restrict__ Vb, __bf16* __restrict__ Ob) {
    extern __shared__ char smem[];
    __bf16* Qs = (__bf16*)smem;                      // [64][512]
    __bf16* Ks = (__bf16*)(smem + 65536);            // [32][512]
    __bf16* Vt = (__bf16*)(smem + 65536 + 32768);    // [512][32] (h-major)
    __bf16* Pb = (__bf16*)(smem + 65536 + 65536);    // per-wave [16][32]

    const int tid  = threadIdx.x;
    const int lane = tid & 31;
    const int wv   = tid >> 5;
    const int n    = lane & 15;
    const int half = lane >> 4;                      // 0 or 1

    const size_t qbase  = ((size_t)blockIdx.y * SEQ + blockIdx.x * 64) * HID;
    const size_t kvbase = (size_t)blockIdx.y * SEQ * HID;

    // stage the 64-row Q tile once (contiguous, async DMA when available)
    for (int i = tid * 8; i < 64 * HID; i += 128 * 8) cp16(Qs + i, Qb + qbase + i);

    v8f acc[32];
#pragma unroll
    for (int t = 0; t < 32; ++t) acc[t] = zero8();
    float m[8], l[8];
#pragma unroll
    for (int r = 0; r < 8; ++r) { m[r] = -1e30f; l[r] = 0.0f; }

    async_join();
    __syncthreads();

    const float scale = 0.044194173824159216f;       // 1/sqrt(512)
    const __bf16* Aq = Qs + wv * 16 * HID;
    __bf16* P = Pb + wv * 16 * 32;

    for (int kt = 0; kt < SEQ / 32; ++kt) {
        // stage 32-row K tile (contiguous, async DMA when available)
        {
            const __bf16* src = Kb + kvbase + (size_t)kt * 32 * HID;
            for (int i = tid * 8; i < 32 * HID; i += 128 * 8) cp16(Ks + i, src + i);
        }
        // stage 32-row V tile transposed -> Vt[h][kv] (VGPR path: transpose)
        {
            const __bf16* src = Vb + kvbase + (size_t)kt * 32 * HID;
            for (int i = tid; i < 32 * HID / 8; i += 128) {
                const int kv = i >> 6;
                const int h0 = (i & 63) * 8;
                const uint4 d = *(const uint4*)(src + (size_t)kv * HID + h0);
                const __bf16* dv = (const __bf16*)&d;
#pragma unroll
                for (int j = 0; j < 8; ++j) Vt[(h0 + j) * 32 + kv] = dv[j];
            }
        }
        async_join();
        __syncthreads();

        // prefetch next K/V tiles into cache while we compute on this one
        if (kt + 1 < SEQ / 32) {
            const __bf16* nk = Kb + kvbase + (size_t)(kt + 1) * 32 * HID;
            const __bf16* nv = Vb + kvbase + (size_t)(kt + 1) * 32 * HID;
            __builtin_prefetch(nk + (size_t)tid * 128, 0, 1);
            __builtin_prefetch(nv + (size_t)tid * 128, 0, 1);
        }

        // ---- scores: 16 q-rows x 32 kv-cols, K=512 ----
        v8f c0 = zero8(), c1 = zero8();
#pragma unroll
        for (int kc = 0; kc < 16; ++kc) {
            const v16bf a  = load_a_frag(Aq + kc * 32, HID, lane);
            const v16bf b0 = load_b_frag(Ks + (size_t)n * HID + kc * 32 + half * 16);
            const v16bf b1 = load_b_frag(Ks + (size_t)(n + 16) * HID + kc * 32 + half * 16);
            c0 = wmma_bf16(a, b0, c0);
            c1 = wmma_bf16(a, b1, c1);
        }

        // ---- online softmax (rows live in 16-lane halves of the wave) ----
#pragma unroll
        for (int r = 0; r < 8; ++r) {
            const float s0 = c0[r] * scale;
            const float s1 = c1[r] * scale;
            float mx = fmaxf(s0, s1);
            mx = fmaxf(mx, __shfl_xor(mx, 1));
            mx = fmaxf(mx, __shfl_xor(mx, 2));
            mx = fmaxf(mx, __shfl_xor(mx, 4));
            mx = fmaxf(mx, __shfl_xor(mx, 8));
            const float mnew = fmaxf(m[r], mx);
            const float p0 = __expf(s0 - mnew);
            const float p1 = __expf(s1 - mnew);
            float rs = p0 + p1;
            rs += __shfl_xor(rs, 1);
            rs += __shfl_xor(rs, 2);
            rs += __shfl_xor(rs, 4);
            rs += __shfl_xor(rs, 8);
            const float alpha = __expf(m[r] - mnew);
            l[r] = l[r] * alpha + rs;
            m[r] = mnew;
#pragma unroll
            for (int t = 0; t < 32; ++t) acc[t][r] *= alpha;
            const int prow = r + half * 8;           // C-layout row of this lane
            P[prow * 32 + n]      = (__bf16)p0;
            P[prow * 32 + n + 16] = (__bf16)p1;
        }

        // ---- O += P @ V  (P re-read from LDS in A-fragment layout) ----
        const v16bf ap = load_a_frag(P, 32, lane);
#pragma unroll
        for (int t = 0; t < 32; ++t) {
            const v16bf bv = load_b_frag(Vt + (t * 16 + n) * 32 + half * 16);
            acc[t] = wmma_bf16(ap, bv, acc[t]);
        }
        __syncthreads();
    }

    // finalize: divide by row sums, store bf16
    const size_t orow0 = (size_t)blockIdx.y * SEQ + blockIdx.x * 64 + wv * 16 + half * 8;
#pragma unroll
    for (int r = 0; r < 8; ++r) {
        const float inv = 1.0f / l[r];
#pragma unroll
        for (int t = 0; t < 32; ++t)
            Ob[(orow0 + r) * HID + t * 16 + n] = (__bf16)(acc[t][r] * inv);
    }
}

// ---------------------------------------------------------------------------
// Kernel 3: proj = O @ Wp + bp ; x = rgb + gate*proj ; LayerNorm(400) -> fp32
// grid: M/64, block 128 (4 waves, 16 rows x 400 cols each)
// ---------------------------------------------------------------------------
__global__ __launch_bounds__(128, 1) void proj_ln_kernel(
    const __bf16* __restrict__ Ob, const float* __restrict__ Wp,
    const float* __restrict__ bp, const float* __restrict__ rgb,
    const float* __restrict__ gate, const float* __restrict__ gamma,
    const float* __restrict__ beta, float* __restrict__ out) {
    extern __shared__ char smem[];
    __bf16* As = (__bf16*)smem;              // [64][32]
    __bf16* Bs = (__bf16*)(smem + 4096);     // [400][32] n-major

    const int tid  = threadIdx.x;
    const int lane = tid & 31;
    const int wv   = tid >> 5;
    const int n    = lane & 15;
    const int half = lane >> 4;
    const int row0 = blockIdx.x * 64;

    v8f acc[25];
#pragma unroll
    for (int t = 0; t < 25; ++t) acc[t] = zero8();

    for (int ks = 0; ks < HID / 32; ++ks) {
        const int k0 = ks * 32;
        // O tile (bf16 source): async DMA into LDS
        for (int c = tid; c < 256; c += 128) {
            const int r = c >> 2, ch = c & 3;
            cp16(As + r * 32 + ch * 8, Ob + (size_t)(row0 + r) * HID + k0 + ch * 8);
        }
        // Wp tile (fp32 -> bf16), coalesced reads of 8 columns, n-major stores
        for (int c = tid; c < 1600; c += 128) {
            const int kk = c & 31;
            const int n0 = (c >> 5) * 8;
            const int k  = k0 + kk;
            const float4 f0 = *(const float4*)(Wp + (size_t)k * DIM + n0);
            const float4 f1 = *(const float4*)(Wp + (size_t)k * DIM + n0 + 4);
            Bs[(n0 + 0) * 32 + kk] = (__bf16)f0.x;
            Bs[(n0 + 1) * 32 + kk] = (__bf16)f0.y;
            Bs[(n0 + 2) * 32 + kk] = (__bf16)f0.z;
            Bs[(n0 + 3) * 32 + kk] = (__bf16)f0.w;
            Bs[(n0 + 4) * 32 + kk] = (__bf16)f1.x;
            Bs[(n0 + 5) * 32 + kk] = (__bf16)f1.y;
            Bs[(n0 + 6) * 32 + kk] = (__bf16)f1.z;
            Bs[(n0 + 7) * 32 + kk] = (__bf16)f1.w;
        }
        async_join();
        __syncthreads();

        const v16bf a = load_a_frag(As + wv * 16 * 32, 32, lane);
#pragma unroll
        for (int t = 0; t < 25; ++t) {
            const v16bf b = load_b_frag(Bs + (t * 16 + n) * 32 + half * 16);
            acc[t] = wmma_bf16(a, b, acc[t]);
        }
        __syncthreads();
    }

    const float g = gate[0];
    const int rbase = row0 + wv * 16 + half * 8;

    // x = rgb + gate * (proj + bias), stored back into acc
#pragma unroll
    for (int t = 0; t < 25; ++t) {
        const int col = t * 16 + n;
        const float b = bp[col];
#pragma unroll
        for (int r = 0; r < 8; ++r)
            acc[t][r] = rgb[(size_t)(rbase + r) * DIM + col] + g * (acc[t][r] + b);
    }

#pragma unroll
    for (int r = 0; r < 8; ++r) {
        float s = 0.0f;
#pragma unroll
        for (int t = 0; t < 25; ++t) s += acc[t][r];
        s += __shfl_xor(s, 1);
        s += __shfl_xor(s, 2);
        s += __shfl_xor(s, 4);
        s += __shfl_xor(s, 8);
        const float mu = s * (1.0f / DIM);
        float vs = 0.0f;
#pragma unroll
        for (int t = 0; t < 25; ++t) {
            const float d = acc[t][r] - mu;
            vs += d * d;
        }
        vs += __shfl_xor(vs, 1);
        vs += __shfl_xor(vs, 2);
        vs += __shfl_xor(vs, 4);
        vs += __shfl_xor(vs, 8);
        const float rstd = rsqrtf(vs * (1.0f / DIM) + 1e-5f);
#pragma unroll
        for (int t = 0; t < 25; ++t) {
            const int col = t * 16 + n;
            out[(size_t)(rbase + r) * DIM + col] =
                (acc[t][r] - mu) * rstd * gamma[col] + beta[col];
        }
    }
}

// ---------------------------------------------------------------------------
extern "C" void kernel_launch(void* const* d_in, const int* in_sizes, int n_in,
                              void* d_out, int out_size, void* d_ws, size_t ws_size,
                              hipStream_t stream) {
    const float* rgb   = (const float*)d_in[0];
    const float* pose  = (const float*)d_in[1];
    const float* Wq    = (const float*)d_in[2];
    const float* bq    = (const float*)d_in[3];
    const float* Wk    = (const float*)d_in[4];
    const float* bk    = (const float*)d_in[5];
    const float* Wv    = (const float*)d_in[6];
    const float* bv    = (const float*)d_in[7];
    const float* Wp    = (const float*)d_in[8];
    const float* bp    = (const float*)d_in[9];
    const float* gamma = (const float*)d_in[10];
    const float* beta  = (const float*)d_in[11];
    const float* gate  = (const float*)d_in[12];

    const size_t M  = (size_t)BATCH * SEQ;       // 65536 rows
    const size_t NE = M * HID;                   // elems per bf16 matrix
    __bf16* Qb = (__bf16*)d_ws;
    __bf16* Kb = Qb + NE;
    __bf16* Vb = Kb + NE;
    __bf16* Ob = Vb + NE;                        // 4 * 64 MB = 256 MB workspace

    const dim3 g1(M / 64, HID / 128, 1);         // 1024 x 4
    const size_t sh1 = (64 * 32 + 128 * 32) * sizeof(__bf16);        // 12 KB
    qkv_kernel<<<g1, 128, sh1, stream>>>(rgb,  Wq, bq, Qb);
    qkv_kernel<<<g1, 128, sh1, stream>>>(pose, Wk, bk, Kb);
    qkv_kernel<<<g1, 128, sh1, stream>>>(pose, Wv, bv, Vb);

    const dim3 g2(SEQ / 64, BATCH, 1);           // 32 x 32
    const size_t sh2 = (64 * HID + 32 * HID + HID * 32 + 4 * 16 * 32) * sizeof(__bf16); // 132 KB
    attn_kernel<<<g2, 128, sh2, stream>>>(Qb, Kb, Vb, Ob);

    const dim3 g3(M / 64, 1, 1);                 // 1024
    const size_t sh3 = (64 * 32 + 400 * 32) * sizeof(__bf16);        // 29 KB
    proj_ln_kernel<<<g3, 128, sh3, stream>>>(Ob, Wp, bp, rgb, gate, gamma, beta,
                                             (float*)d_out);
}
